// W8A8BFP32OFP32LinearWithQuantScale_71725953843874
// MI455X (gfx1250) — 8-bit matrix operand order repaired and checked by reading, compiled, not run on hardware
//
#include <hip/hip_runtime.h>
#include <cstdint>

typedef __attribute__((ext_vector_type(8))) int   v8i_t;
typedef __attribute__((ext_vector_type(4))) int   i4_t;
typedef __attribute__((ext_vector_type(2))) int   i2_t;
typedef __attribute__((ext_vector_type(4))) float f4_t;

#define AS1 __attribute__((address_space(1)))
#define AS3 __attribute__((address_space(3)))

#define BM 128
#define BN 256
#define BK 64
#define LDT 80

#if __has_builtin(__builtin_amdgcn_global_load_async_to_lds_b128)
#define USE_ASYNC_COPY 1
#else
#define USE_ASYNC_COPY 0
#endif

__global__ __launch_bounds__(256) void quant_x_kernel(
    const float* __restrict__ x, const float* __restrict__ qscale,
    int8_t* __restrict__ xq, long long n4)
{
  long long i = (long long)blockIdx.x * blockDim.x + threadIdx.x;
  if (i >= n4) return;
  const float inv = 1.0f / qscale[0];
  f4_t v = ((const f4_t*)x)[i];
  int q0 = (int)rintf(fminf(fmaxf(v[0] * inv, -128.0f), 127.0f));
  int q1 = (int)rintf(fminf(fmaxf(v[1] * inv, -128.0f), 127.0f));
  int q2 = (int)rintf(fminf(fmaxf(v[2] * inv, -128.0f), 127.0f));
  int q3 = (int)rintf(fminf(fmaxf(v[3] * inv, -128.0f), 127.0f));
  ((int*)xq)[i] = (q0 & 255) | ((q1 & 255) << 8) | ((q2 & 255) << 16) | ((q3 & 255) << 24);
}

__global__ __launch_bounds__(256) void pack_w_kernel(
    const int* __restrict__ w, int8_t* __restrict__ wq, long long n4)
{
  long long i = (long long)blockIdx.x * blockDim.x + threadIdx.x;
  if (i >= n4) return;
  i4_t v = ((const i4_t*)w)[i];
  ((int*)wq)[i] = (v[0] & 255) | ((v[1] & 255) << 8) | ((v[2] & 255) << 16) | ((v[3] & 255) << 24);
}

__global__ __launch_bounds__(256) void w8a8_gemm_kernel(
    const int8_t* __restrict__ Aq, const int8_t* __restrict__ Bq,
    const float* __restrict__ wscale, const float* __restrict__ qscale,
    const float* __restrict__ bias, float* __restrict__ out,
    int M, int N, int K)
{
  __shared__ __align__(16) int8_t sA[2][BM * LDT];
  __shared__ __align__(16) int8_t sB[2][BN * LDT];

  const int tid   = threadIdx.x;
  const int lane  = tid & 31;
  const int wave  = tid >> 5;
  const int wm    = wave & 1;
  const int wn    = wave >> 1;
  const int l15   = lane & 15;
  const int lhalf = lane >> 4;
  const int kbA   = lhalf * 16;
  const int kbB   = lhalf * 16;

  const int bm = blockIdx.y * BM;
  const int bn = blockIdx.x * BN;

  const int arow = tid >> 1;
  const int acol = (tid & 1) * 32;

  const int8_t* gA = Aq + (size_t)(bm + arow) * K + acol;
  const int8_t* gB = Bq + (size_t)(bn + tid) * K;
  int8_t* lA0 = &sA[0][arow * LDT + acol];
  int8_t* lA1 = &sA[1][arow * LDT + acol];
  int8_t* lB0 = &sB[0][tid * LDT];
  int8_t* lB1 = &sB[1][tid * LDT];

  auto issue_copy = [&](int k0, int buf) {
    int8_t* lA = buf ? lA1 : lA0;
    int8_t* lB = buf ? lB1 : lB0;
#if USE_ASYNC_COPY
    AS1 i4_t* ga = (AS1 i4_t*)(gA + k0);
    AS1 i4_t* gb = (AS1 i4_t*)(gB + k0);
    AS3 i4_t* la = (AS3 i4_t*)lA;
    AS3 i4_t* lb = (AS3 i4_t*)lB;
    __builtin_amdgcn_global_load_async_to_lds_b128(ga, la, 0, 0);
    __builtin_amdgcn_global_load_async_to_lds_b128(ga, la, 16, 0);
    __builtin_amdgcn_global_load_async_to_lds_b128(gb, lb, 0, 0);
    __builtin_amdgcn_global_load_async_to_lds_b128(gb, lb, 16, 0);
    __builtin_amdgcn_global_load_async_to_lds_b128(gb, lb, 32, 0);
    __builtin_amdgcn_global_load_async_to_lds_b128(gb, lb, 48, 0);
#else
    i4_t a0 = *(const i4_t*)(gA + k0);
    i4_t a1 = *(const i4_t*)(gA + k0 + 16);
    i4_t b0 = *(const i4_t*)(gB + k0);
    i4_t b1 = *(const i4_t*)(gB + k0 + 16);
    i4_t b2 = *(const i4_t*)(gB + k0 + 32);
    i4_t b3 = *(const i4_t*)(gB + k0 + 48);
    *(i4_t*)(lA)      = a0;
    *(i4_t*)(lA + 16) = a1;
    *(i4_t*)(lB)      = b0;
    *(i4_t*)(lB + 16) = b1;
    *(i4_t*)(lB + 32) = b2;
    *(i4_t*)(lB + 48) = b3;
#endif
  };

  v8i_t acc[4][4] = {};

  const int nk = K / BK;
  issue_copy(0, 0);

  for (int kt = 0; kt < nk; ++kt) {
    const int cur = kt & 1;
#if USE_ASYNC_COPY
#if __has_builtin(__builtin_amdgcn_s_wait_asynccnt)
    __builtin_amdgcn_s_wait_asynccnt(0);
#else
    asm volatile("s_wait_asynccnt 0x0" ::: "memory");
#endif
#endif
    __syncthreads();
    if (kt + 1 < nk) issue_copy((kt + 1) * BK, cur ^ 1);

    const int8_t* bufA = sA[cur];
    const int8_t* bufB = sB[cur];

    v8i_t afrag[4];
#pragma unroll
    for (int i = 0; i < 4; ++i) {
      const int8_t* p = bufA + (wm * 64 + i * 16 + l15) * LDT + kbA;
      union { i2_t d[4]; v8i_t v; } u;
      u.d[0] = *(const i2_t*)(p);
      u.d[1] = *(const i2_t*)(p + 8);
      u.d[2] = *(const i2_t*)(p + 32);
      u.d[3] = *(const i2_t*)(p + 40);
      afrag[i] = u.v;
    }

    v8i_t bfrag[4];
#pragma unroll
    for (int j = 0; j < 4; ++j) {
      const int8_t* p = bufB + (wn * 64 + j * 16 + l15) * LDT + kbB;
      union { i4_t d[2]; v8i_t v; } u;
      u.d[0] = *(const i4_t*)(p);
      u.d[1] = *(const i4_t*)(p + 32);
      bfrag[j] = u.v;
    }

#pragma unroll
    for (int i = 0; i < 4; ++i)
#pragma unroll
      for (int j = 0; j < 4; ++j)
        acc[i][j] = __builtin_amdgcn_wmma_i32_16x16x64_iu8(
             true, afrag[i],  true, bfrag[j],
            acc[i][j],  false,  false);
  }

  const float qs = qscale[0];
#pragma unroll
  for (int j = 0; j < 4; ++j) {
    const int n = bn + wn * 64 + j * 16 + l15;
    const float sc = wscale[n] * qs;
    const float bz = bias[n];
#pragma unroll
    for (int i = 0; i < 4; ++i) {
      const int mb = bm + wm * 64 + i * 16 + lhalf * 8;
#pragma unroll
      for (int r = 0; r < 8; ++r) {
        out[(size_t)(mb + r) * N + n] = (float)acc[i][j][r] * sc + bz;
      }
    }
  }
}

extern "C" void kernel_launch(void* const* d_in, const int* in_sizes, int n_in,
                              void* d_out, int out_size, void* d_ws, size_t ws_size,
                              hipStream_t stream) {
  const float* x      = (const float*)d_in[0];
  const int*   w      = (const int*)d_in[1];
  const float* wscale = (const float*)d_in[2];
  const float* qscale = (const float*)d_in[3];
  const float* bias   = (const float*)d_in[4];
  float* out = (float*)d_out;

  const int N = in_sizes[2];
  const int K = in_sizes[1] / N;
  const int M = in_sizes[0] / K;

  int8_t* xq = (int8_t*)d_ws;
  int8_t* wq = xq + (size_t)M * K;

  const long long nx4 = (long long)M * K / 4;
  const long long nw4 = (long long)N * K / 4;

  quant_x_kernel<<<(unsigned)((nx4 + 255) / 256), 256, 0, stream>>>(x, qscale, xq, nx4);
  pack_w_kernel<<<(unsigned)((nw4 + 255) / 256), 256, 0, stream>>>(w, wq, nw4);

  dim3 grid(N / BN, M / BM);
  w8a8_gemm_kernel<<<grid, 256, 0, stream>>>(xq, wq, wscale, qscale, bias, out, M, N, K);
}
